// MaskedBEVTransformer_24721831756413
// MI455X (gfx1250) — matrix operand order repaired and checked by reading, not compiled, not run on hardware
//
#include <hip/hip_runtime.h>
#include <hip/hip_bf16.h>
#include <math.h>

typedef __attribute__((ext_vector_type(16))) _Float16 v16h;
typedef __attribute__((ext_vector_type(2)))  _Float16 v2h;
typedef __attribute__((ext_vector_type(8)))  float    v8f;

#define WMMA_F16(A_, B_, C_) \
  __builtin_amdgcn_wmma_f32_16x16x32_f16(false, (A_), false, (B_), (short)0, (C_), false, false)

constexpr int NR   = 8192;   // RV tokens
constexpr int NB   = 16384;  // BEV cells
constexpr int CB   = 80;     // bev channels
constexpr int CRV  = 384;    // rv channels
constexpr int CA   = 128;    // attention dim
constexpr int MH   = 512;    // mlp hidden
constexpr int MC   = 96;     // candidates per token
constexpr float EPSF = 1e-5f;

// ---------------- WMMA fragment loaders (contiguous k = 16*(lane/16) + s for A and B) ----------------
// A: 16x32 f16 from LDS, row-major (lda in halves). Halves 0..7 / 8..15 are
// contiguous runs of 8 -> 2x ds_load_b128 per fragment.
__device__ inline v16h ld_fragA(const _Float16* __restrict__ tile, int lda, int lane, int kmax) {
  v16h a;
  int row  = lane & 15;
  int k_lo = (lane < 16) ? 0 : 16;
  int k_hi = (lane < 16) ? 8 : 24;
#pragma unroll
  for (int i = 0; i < 8; ++i) {
    int ka = k_lo + i, kb = k_hi + i;
    a[i]     = (ka < kmax) ? tile[row * lda + ka] : (_Float16)0.0f;
    a[i + 8] = (kb < kmax) ? tile[row * lda + kb] : (_Float16)0.0f;
  }
  return a;
}

// B from a K-major (transposed) source: src[col*lds + k]; the 16 K-values per
// lane are contiguous -> one 32B vectorizable LDS read.
__device__ inline v16h ld_fragB_T(const _Float16* __restrict__ src, int lds, int lane) {
  v16h b;
  int col = lane & 15;
  int kb  = (lane < 16) ? 0 : 16;
#pragma unroll
  for (int i = 0; i < 16; ++i) b[i] = src[col * lds + kb + i];
  return b;
}

// B: weights pre-swizzled to fragment layout [tiles][32 lanes][16 halves];
// one aligned 32-byte v16h load per lane -> 2x global_load_b128, coalesced.
__device__ inline v16h ld_fragB_packed(const _Float16* __restrict__ base, int tile, int lane) {
  const v16h* p = (const v16h*)base;
  return p[tile * 32 + lane];
}

__device__ inline float wave_sum(float v) {
#pragma unroll
  for (int m = 16; m > 0; m >>= 1) v += __shfl_xor(v, m, 32);
  return v;
}

// ---------------- prep kernels ----------------
__global__ void k_prep_bev(const float* __restrict__ bev, const float* __restrict__ pos,
                           _Float16* __restrict__ bev16) {
  int i = blockIdx.x * blockDim.x + threadIdx.x;
  if (i >= NB * CB) return;
  int n = i / CB, c = i % CB;                 // [Nb][80] row-major f16
  bev16[i] = (_Float16)(bev[c * NB + n] + pos[c * NB + n]);
}

// Pack W[K x N] (f32, row-major) into B-fragment layout:
// dst[((nt*(Kpad/32) + ks)*32 + lane)*16 + e]; rows >= Ksrc are zero.
__global__ void k_pack_B(const float* __restrict__ src, _Float16* __restrict__ dst,
                         int Ksrc, int Kpad, int N) {
  int i = blockIdx.x * blockDim.x + threadIdx.x;
  int total = (N / 16) * (Kpad / 32) * 32 * 16;
  if (i >= total) return;
  int e    = i & 15;
  int lane = (i >> 4) & 31;
  int t    = i >> 9;
  int nks  = Kpad / 32;
  int nt = t / nks, ks = t % nks;
  int kk  = ks * 32 + ((lane < 16) ? 0 : 16) + e;
  int col = nt * 16 + (lane & 15);
  dst[i] = (kk < Ksrc) ? (_Float16)src[kk * N + col] : (_Float16)0.0f;
}

// ---------------- Q projection: LN(rv) @ Wq + bq ----------------
__global__ void __launch_bounds__(256)
k_qproj(const float* __restrict__ rv, const float* __restrict__ qng,
        const float* __restrict__ qnb, const _Float16* __restrict__ WqP,
        const float* __restrict__ bq, float* __restrict__ Qf32,
        _Float16* __restrict__ Qf16) {
  __shared__ __align__(16) _Float16 sQ[16 * CRV];
  const int tok0 = blockIdx.x * 16;
  const int tid = threadIdx.x, wave = tid >> 5, lane = tid & 31;

  // LayerNorm: 2 rows per wave, 12 elems per lane
#pragma unroll
  for (int rr = 0; rr < 2; ++rr) {
    int r = wave * 2 + rr;
    float v[12]; float s1 = 0.f;
#pragma unroll
    for (int j = 0; j < 12; ++j) { v[j] = rv[(size_t)(tok0 + r) * CRV + j * 32 + lane]; s1 += v[j]; }
    s1 = wave_sum(s1);
    float mean = s1 / 384.f, s2 = 0.f;
#pragma unroll
    for (int j = 0; j < 12; ++j) { float d = v[j] - mean; s2 += d * d; }
    s2 = wave_sum(s2);
    float rstd = rsqrtf(s2 / 384.f + EPSF);
#pragma unroll
    for (int j = 0; j < 12; ++j) {
      int c = j * 32 + lane;
      sQ[r * CRV + c] = (_Float16)((v[j] - mean) * rstd * qng[c] + qnb[c]);
    }
  }
  __syncthreads();

  // [16,384] @ [384,128]: wave w owns N-tile w; 12 K-steps, nks = 12
  __builtin_prefetch(WqP + wave * 12 * 512, 0, 0);
  v8f acc = {};
#pragma unroll
  for (int k = 0; k < 12; ++k) {
    v16h A = ld_fragA(sQ + k * 32, CRV, lane, 32);
    v16h B = ld_fragB_packed(WqP, wave * 12 + k, lane);
    acc = WMMA_F16(A, B, acc);
  }
  int col = lane & 15, rb = (lane < 16) ? 0 : 8;
  int cg = wave * 16 + col;
  float bb = bq[cg];
#pragma unroll
  for (int i = 0; i < 8; ++i) {
    float v = acc[i] + bb;
    Qf32[(size_t)(tok0 + rb + i) * CA + cg] = v;
    Qf16[(size_t)(tok0 + rb + i) * CA + cg] = (_Float16)v;
  }
}

// ---------------- attention: gather + K/V + logits + attn@V all on WMMA ----------------
__global__ void __launch_bounds__(256)
k_attn(const _Float16* __restrict__ bev16,
       const _Float16* __restrict__ WkP, const float* __restrict__ bk,
       const _Float16* __restrict__ WvP, const float* __restrict__ bv,
       const float* __restrict__ Qf32, const _Float16* __restrict__ Qf16,
       const int* __restrict__ cand, const unsigned char* __restrict__ pad,
       const float* __restrict__ ln1g, const float* __restrict__ ln1b,
       float* __restrict__ Xf32) {
  __shared__ __align__(16) _Float16 sG[MC * CB];   // gathered f16; reused as f32 scratch
  __shared__ __align__(16) _Float16 sK[MC * CA];   // K, row-major [m][c]
  __shared__ __align__(16) _Float16 sVt[CA * MC];  // V transposed [c][m]
  __shared__ int           sIdx[MC];
  __shared__ unsigned char sMask[MC];
  __shared__ __align__(16) _Float16 sQh[CA];       // this token's Q, f16
  const int n = blockIdx.x;
  const int tid = threadIdx.x, wave = tid >> 5, lane = tid & 31;
  const int col = lane & 15, rb = (lane < 16) ? 0 : 8;

  if (tid < MC) {
    sIdx[tid]  = cand[n * MC + tid];
    sMask[tid] = pad[n * MC + tid];
  }
  if (tid < CA) sQh[tid] = Qf16[(size_t)n * CA + tid];
  __syncthreads();

  // dword gather of 96 candidate rows (80 halves each)
  const unsigned int* bevu = (const unsigned int*)bev16;
  unsigned int* sGu = (unsigned int*)sG;
  for (int i = tid; i < MC * (CB / 2); i += 256) {
    int m = i / (CB / 2), c2 = i % (CB / 2);
    int idx = sIdx[m];
    sGu[m * (CB / 2) + c2] = (idx < 0) ? 0u : bevu[(size_t)idx * (CB / 2) + c2];
  }
  __syncthreads();

  // K = G@Wk + bk (row-major), Vt = (G@Wv + bv)^T : 96 tiles over 8 waves
  __builtin_prefetch(WkP, 0, 0);
  __builtin_prefetch(WvP, 0, 0);
  for (int t = wave; t < 96; t += 8) {
    int which = t / 48;
    int tt = t % 48;
    int mtile = tt >> 3, ntile = tt & 7;
    const _Float16* Wg = which ? WvP : WkP;
    const float* bias  = which ? bv : bk;
    v8f acc = {};
#pragma unroll
    for (int k = 0; k < 3; ++k) {
      v16h A = ld_fragA(sG + mtile * 16 * CB + k * 32, CB, lane, CB - k * 32); // tail zeroed
      v16h B = ld_fragB_packed(Wg, ntile * 3 + k, lane);                       // nks = 3
      acc = WMMA_F16(A, B, acc);
    }
    int cg = ntile * 16 + col;
    float bc = bias[cg];
    if (which) {
      // transposed store: contiguous 8 halves per lane -> merges to ds_store_b128
#pragma unroll
      for (int i = 0; i < 8; ++i)
        sVt[cg * MC + mtile * 16 + rb + i] = (_Float16)(acc[i] + bc);
    } else {
#pragma unroll
      for (int i = 0; i < 8; ++i)
        sK[(mtile * 16 + rb + i) * CA + cg] = (_Float16)(acc[i] + bc);
    }
  }
  __syncthreads();

  // logits[m][h] = sum_c K[m][c] * Qdiag[c][h]  (block-diagonal Q), WMMA over K=128.
  float* sLog = (float*)sG;   // [4][96] attn weights
  float* sX   = sLog + 4 * MC;
  float* sSt  = sX + CA;
  const float scale = 0.17677669529663687f;  // 1/sqrt(32)
  if (wave < 6) {                            // one m-tile per wave
    int mtile = wave;
    v8f acc = {};
#pragma unroll
    for (int k = 0; k < 4; ++k) {
      v16h A = ld_fragA(sK + mtile * 16 * CA + k * 32, CA, lane, 32);
      // B[c][h] = Q[c] iff c>>5 == h; per-lane-uniform select per k-step
      bool on = (col == k);
      const _Float16* qp = sQh + k * 32 + ((lane < 16) ? 0 : 16);
      v16h B;
#pragma unroll
      for (int i = 0; i < 16; ++i) B[i] = on ? qp[i] : (_Float16)0.0f;
      acc = WMMA_F16(A, B, acc);
    }
    if (col < 4) {                           // cols 0..3 hold heads
#pragma unroll
      for (int i = 0; i < 8; ++i) {
        int m = mtile * 16 + rb + i;
        sLog[col * MC + m] = sMask[m] ? -1e30f : acc[i] * scale;
      }
    }
  }
  __syncthreads();

  // softmax: one wave per head, 3 elements per lane
  if (wave < 4) {
    int h = wave;
    float v0 = sLog[h * MC + lane], v1 = sLog[h * MC + lane + 32], v2 = sLog[h * MC + lane + 64];
    float mx = fmaxf(v0, fmaxf(v1, v2));
#pragma unroll
    for (int m = 16; m > 0; m >>= 1) mx = fmaxf(mx, __shfl_xor(mx, m, 32));
    float e0 = __expf(v0 - mx), e1 = __expf(v1 - mx), e2 = __expf(v2 - mx);
    float s = wave_sum(e0 + e1 + e2);
    float inv = 1.f / s;
    sLog[h * MC + lane]      = e0 * inv;
    sLog[h * MC + lane + 32] = e1 * inv;
    sLog[h * MC + lane + 64] = e2 * inv;
  }
  __syncthreads();

  // O[h][c] = sum_m attn[h][m] * V[m][c] via WMMA; wave w owns column tile w.
  {
    int nt = wave;
    v8f acc = {};
#pragma unroll
    for (int k = 0; k < 3; ++k) {
      // A rows 0..3 = attn heads (f32 -> f16), rows 4..15 zero
      v16h A = {};
      int row = lane & 15;
      if (row < 4) {
        int k_lo = (lane < 16) ? 0 : 16;
        int k_hi = (lane < 16) ? 8 : 24;
#pragma unroll
        for (int i = 0; i < 8; ++i) {
          A[i]     = (_Float16)sLog[row * MC + k * 32 + k_lo + i];
          A[i + 8] = (_Float16)sLog[row * MC + k * 32 + k_hi + i];
        }
      }
      v16h B = ld_fragB_T(sVt + (nt * 16) * MC + k * 32, MC, lane);
      acc = WMMA_F16(A, B, acc);
    }
    // col c = nt*16+lane (lanes 0..15); its head h = nt>>1 lives in acc[h]
    if (lane < 16) {
      int c = nt * 16 + lane;
      sX[c] = acc[nt >> 1] + Qf32[(size_t)n * CA + c];
    }
  }
  __syncthreads();

  // LN1
  if (wave == 0) {
    float s1 = 0.f, s2 = 0.f;
#pragma unroll
    for (int j = 0; j < 4; ++j) { float x = sX[j * 32 + lane]; s1 += x; s2 += x * x; }
    s1 = wave_sum(s1); s2 = wave_sum(s2);
    if (lane == 0) {
      float mean = s1 / 128.f;
      float var  = s2 / 128.f - mean * mean;
      sSt[0] = mean; sSt[1] = rsqrtf(var + EPSF);
    }
  }
  __syncthreads();
  if (tid < CA)
    Xf32[(size_t)n * CA + tid] = (sX[tid] - sSt[0]) * sSt[1] * ln1g[tid] + ln1b[tid];
}

// ---------------- MLP + output projection ----------------
__global__ void __launch_bounds__(256)
k_mlp_out(const float* __restrict__ Xf32,
          const float* __restrict__ ln2g, const float* __restrict__ ln2b,
          const _Float16* __restrict__ W1P, const float* __restrict__ b1,
          const _Float16* __restrict__ W2P, const float* __restrict__ b2,
          const _Float16* __restrict__ WoP, const float* __restrict__ bo,
          float* __restrict__ out) {
  __shared__ __align__(16) _Float16 sA[16 * CA];
  __shared__ __align__(16) _Float16 sH[16 * MH];
  __shared__ __align__(16) _Float16 sX2[16 * CA];
  const int tok0 = blockIdx.x * 16;
  const int tid = threadIdx.x, wave = tid >> 5, lane = tid & 31;
  const int col = lane & 15, rb = (lane < 16) ? 0 : 8;

  // LN2 of x: 2 rows per wave
#pragma unroll
  for (int rr = 0; rr < 2; ++rr) {
    int r = wave * 2 + rr;
    float v[4]; float s1 = 0.f;
#pragma unroll
    for (int j = 0; j < 4; ++j) { v[j] = Xf32[(size_t)(tok0 + r) * CA + j * 32 + lane]; s1 += v[j]; }
    s1 = wave_sum(s1);
    float mean = s1 / 128.f, s2 = 0.f;
#pragma unroll
    for (int j = 0; j < 4; ++j) { float d = v[j] - mean; s2 += d * d; }
    s2 = wave_sum(s2);
    float rstd = rsqrtf(s2 / 128.f + EPSF);
#pragma unroll
    for (int j = 0; j < 4; ++j) {
      int c = j * 32 + lane;
      sA[r * CA + c] = (_Float16)((v[j] - mean) * rstd * ln2g[c] + ln2b[c]);
    }
  }
  __syncthreads();

  // H = gelu(A @ W1 + b1)   [16,128]@[128,512], nks = 4
  __builtin_prefetch(W1P + wave * 4 * 512, 0, 0);
  for (int nt = wave; nt < 32; nt += 8) {
    v8f acc = {};
#pragma unroll
    for (int k = 0; k < 4; ++k) {
      v16h A = ld_fragA(sA + k * 32, CA, lane, 32);
      v16h B = ld_fragB_packed(W1P, nt * 4 + k, lane);
      acc = WMMA_F16(A, B, acc);
    }
    int cg = nt * 16 + col;
    float bb = b1[cg];
#pragma unroll
    for (int i = 0; i < 8; ++i) {
      float h = acc[i] + bb;
      float g = 0.5f * h * (1.0f + erff(h * 0.70710678118f));   // exact gelu
      sH[(rb + i) * MH + cg] = (_Float16)g;
    }
  }
  __syncthreads();

  // X2 = H @ W2 + b2 + X    [16,512]@[512,128], nks = 16
  {
    int nt = wave;
    v8f acc = {};
#pragma unroll
    for (int k = 0; k < 16; ++k) {
      v16h A = ld_fragA(sH + k * 32, MH, lane, 32);
      v16h B = ld_fragB_packed(W2P, nt * 16 + k, lane);
      acc = WMMA_F16(A, B, acc);
    }
    int cg = nt * 16 + col;
    float bb = b2[cg];
#pragma unroll
    for (int i = 0; i < 8; ++i) {
      int r = rb + i;
      float v = acc[i] + bb + Xf32[(size_t)(tok0 + r) * CA + cg];
      sX2[r * CA + cg] = (_Float16)v;
    }
  }
  __syncthreads();

  // OUT = X2 @ Wo + bo      [16,128]@[128,384], nks = 4
  for (int nt = wave; nt < 24; nt += 8) {
    v8f acc = {};
#pragma unroll
    for (int k = 0; k < 4; ++k) {
      v16h A = ld_fragA(sX2 + k * 32, CA, lane, 32);
      v16h B = ld_fragB_packed(WoP, nt * 4 + k, lane);
      acc = WMMA_F16(A, B, acc);
    }
    int cg = nt * 16 + col;
    float bb = bo[cg];
#pragma unroll
    for (int i = 0; i < 8; ++i)
      out[(size_t)(tok0 + rb + i) * CRV + cg] = acc[i] + bb;
  }
}

// ---------------- host launch ----------------
extern "C" void kernel_launch(void* const* d_in, const int* in_sizes, int n_in,
                              void* d_out, int out_size, void* d_ws, size_t ws_size,
                              hipStream_t stream) {
  (void)in_sizes; (void)n_in; (void)out_size; (void)ws_size;
  const float* rv   = (const float*)d_in[0];
  const float* bev  = (const float*)d_in[1];
  const int*   cand = (const int*)d_in[2];
  const unsigned char* pad = (const unsigned char*)d_in[3];  // jnp bool = 1 byte
  const float* pos  = (const float*)d_in[4];
  const float* qng  = (const float*)d_in[5];
  const float* qnb  = (const float*)d_in[6];
  const float* Wq   = (const float*)d_in[7];
  const float* bq   = (const float*)d_in[8];
  const float* Wk   = (const float*)d_in[9];
  const float* bk   = (const float*)d_in[10];
  const float* Wv   = (const float*)d_in[11];
  const float* bv   = (const float*)d_in[12];
  const float* ln1g = (const float*)d_in[13];
  const float* ln1b = (const float*)d_in[14];
  const float* ln2g = (const float*)d_in[15];
  const float* ln2b = (const float*)d_in[16];
  const float* W1   = (const float*)d_in[17];
  const float* b1   = (const float*)d_in[18];
  const float* W2   = (const float*)d_in[19];
  const float* b2   = (const float*)d_in[20];
  const float* Wo   = (const float*)d_in[21];
  const float* bo   = (const float*)d_in[22];
  float* out = (float*)d_out;

  char* ws = (char*)d_ws;
  size_t off = 0;
  auto wsalloc = [&](size_t bytes) -> char* {
    char* p = ws + off;
    off += (bytes + 255) & ~(size_t)255;
    return p;
  };
  _Float16* bev16 = (_Float16*)wsalloc((size_t)NB * CB * 2);
  _Float16* WqP   = (_Float16*)wsalloc((size_t)CRV * CA * 2);   // packed, Kpad=384
  _Float16* WkP   = (_Float16*)wsalloc((size_t)96 * CA * 2);    // packed, Kpad=96
  _Float16* WvP   = (_Float16*)wsalloc((size_t)96 * CA * 2);
  _Float16* W1P   = (_Float16*)wsalloc((size_t)CA * MH * 2);    // packed, Kpad=128
  _Float16* W2P   = (_Float16*)wsalloc((size_t)MH * CA * 2);    // packed, Kpad=512
  _Float16* WoP   = (_Float16*)wsalloc((size_t)CA * CRV * 2);   // packed, Kpad=128
  float*    Qf32  = (float*)wsalloc((size_t)NR * CA * 4);
  _Float16* Qf16  = (_Float16*)wsalloc((size_t)NR * CA * 2);
  float*    Xf32  = (float*)wsalloc((size_t)NR * CA * 4);

  k_prep_bev<<<(NB * CB + 255) / 256, 256, 0, stream>>>(bev, pos, bev16);
  k_pack_B<<<(CRV * CA + 255) / 256, 256, 0, stream>>>(Wq, WqP, 384, 384, CA);
  k_pack_B<<<(96 * CA + 255) / 256, 256, 0, stream>>>(Wk, WkP, 80, 96, CA);
  k_pack_B<<<(96 * CA + 255) / 256, 256, 0, stream>>>(Wv, WvP, 80, 96, CA);
  k_pack_B<<<(CA * MH + 255) / 256, 256, 0, stream>>>(W1, W1P, 128, 128, MH);
  k_pack_B<<<(MH * CA + 255) / 256, 256, 0, stream>>>(W2, W2P, 512, 512, CA);
  k_pack_B<<<(CA * CRV + 255) / 256, 256, 0, stream>>>(Wo, WoP, 128, 128, CRV);

  k_qproj<<<NR / 16, 256, 0, stream>>>(rv, qng, qnb, WqP, bq, Qf32, Qf16);
  k_attn<<<NR, 256, 0, stream>>>(bev16, WkP, bk, WvP, bv, Qf32, Qf16, cand, pad,
                                 ln1g, ln1b, Xf32);
  k_mlp_out<<<NR / 16, 256, 0, stream>>>(Xf32, ln2g, ln2b, W1P, b1, W2P, b2,
                                         WoP, bo, out);
}
